// CrossAttentionLayer_1984274891435
// MI455X (gfx1250) — matrix operand order repaired and checked by reading, not compiled, not run on hardware
//
#include <hip/hip_runtime.h>
#include <hip/hip_bf16.h>

typedef __attribute__((ext_vector_type(16))) _Float16 v16h;
typedef __attribute__((ext_vector_type(8)))  _Float16 v8h;
typedef __attribute__((ext_vector_type(8)))  float    v8f;

// Problem constants
static constexpr int Bsz   = 8;
static constexpr int Nseq  = 2048;
static constexpr int Jseq  = 1024;
static constexpr int QD    = 1024;
static constexpr int CD    = 768;
static constexpr int Hh    = 16;
static constexpr int Dd    = 64;
static constexpr int INNER = 1024;

static __device__ __forceinline__ v8f wmma32(v16h a, v16h b, v8f c) {
  // D = A(16x32 f16) * B(32x16 f16) + C(16x16 f32)
  return __builtin_amdgcn_wmma_f32_16x16x32_f16(false, a, false, b, (short)0, c,
                                                false, false);
}

static __device__ __forceinline__ v16h ldfrag(const _Float16* p0, const _Float16* p1) {
  v8h a = *(const v8h*)p0;
  v8h b = *(const v8h*)p1;
  return __builtin_shufflevector(a, b, 0,1,2,3,4,5,6,7,8,9,10,11,12,13,14,15);
}

// ---------------------------------------------------------------------------
// Tiled GEMM: C[M,Nc] = A[M,K] x B[K,Nc]  (WMMA f16 in, f32 accum)
// Block tile 128x128, BK=32, 256 threads = 8 waves (4x2), wave tile 32x64.
// Per wave per K-step: 8 WMMAs from 2 A-frags + 4 B-frags (0.75KB DS/WMMA).
// LDS stride 56 halves = 112B = 28 dwords -> conflict-free (gcd(28,64)=4,
// 28*ln mod 64 distinct for ln=0..15) and 16B-aligned rows.
// A_F16: A operand already f16 (else f32, converted at LDS fill).
// OUT_F32_BIAS: write f32 + bias, else write f16.
// ---------------------------------------------------------------------------
template<bool A_F16, bool OUT_F32_BIAS>
__global__ __launch_bounds__(256)
void gemm_kernel(const void* __restrict__ Aptr, const float* __restrict__ Bptr,
                 const float* __restrict__ bias, void* __restrict__ Cptr,
                 int M, int K, int Nc)
{
  constexpr int BM = 128, BN = 128, BK = 32, LDT = 56;
  __shared__ __align__(16) _Float16 sA[BM * LDT];     // [m][k]  14KB
  __shared__ __align__(16) _Float16 sB[BN * LDT];     // [n][k]  14KB (transposed)

  const int tid  = threadIdx.x;
  const int lane = tid & 31;
  const int wave = tid >> 5;
  const int wm   = wave >> 1;   // 0..3 -> 32 rows each
  const int wn   = wave & 1;    // 0..1 -> 64 cols each
  const int s    = lane >> 4;   // lane half select
  const int ln   = lane & 15;

  const int rowBase = blockIdx.y * BM;
  const int colBase = blockIdx.x * BN;

  v8f acc[2][4];
#pragma unroll
  for (int i = 0; i < 2; ++i)
#pragma unroll
    for (int j = 0; j < 4; ++j) acc[i][j] = {};

  const int nK = K / BK;
  for (int kk = 0; kk < nK; ++kk) {
    const int kBase = kk * BK;
    // ---- fill A tile: 128x32, 16 halves per thread (contiguous) ----
    {
      const int r = tid >> 1;
      const int c = (tid & 1) * 16;
      v8h lo, hi;
      if (A_F16) {
        const _Float16* ap = (const _Float16*)Aptr + (size_t)(rowBase + r) * K + kBase + c;
        lo = *(const v8h*)ap;
        hi = *(const v8h*)(ap + 8);
      } else {
        const float* ap = (const float*)Aptr + (size_t)(rowBase + r) * K + kBase + c;
#pragma unroll
        for (int i = 0; i < 8; ++i) lo[i] = (_Float16)ap[i];
#pragma unroll
        for (int i = 0; i < 8; ++i) hi[i] = (_Float16)ap[8 + i];
      }
      *(v8h*)(sA + r * LDT + c)     = lo;
      *(v8h*)(sA + r * LDT + c + 8) = hi;
    }
    // ---- fill B tile 32x128, stored transposed [n][k] ----
    {
      const int kr = tid >> 3;            // 0..31
      const int c  = (tid & 7) * 16;      // 0..112
      const float* bp = Bptr + (size_t)(kBase + kr) * Nc + colBase + c;
#pragma unroll
      for (int i = 0; i < 16; ++i) sB[(c + i) * LDT + kr] = (_Float16)bp[i];
    }
    __syncthreads();

    // ---- fragments (ISA 16-bit A 16x32 / B 32x16 lane layouts) ----
    v16h af[2], bf[4];
#pragma unroll
    for (int mt = 0; mt < 2; ++mt) {
      const _Float16* p = sA + (wm * 32 + mt * 16 + ln) * LDT + s * 8;
      af[mt] = ldfrag(p, p + 16);        // K = s*8+h (h<8), 16+s*8+(h-8)
    }
#pragma unroll
    for (int nt = 0; nt < 4; ++nt) {
      const _Float16* p = sB + (wn * 64 + nt * 16 + ln) * LDT + s * 8;
      bf[nt] = ldfrag(p, p + 16);        // K = s*8+h (h<8), 16+s*8+(h-8)
    }
#pragma unroll
    for (int mt = 0; mt < 2; ++mt)
#pragma unroll
      for (int nt = 0; nt < 4; ++nt)
        acc[mt][nt] = wmma32(af[mt], bf[nt], acc[mt][nt]);
    __syncthreads();
  }

  // ---- epilogue: C/D layout row M = v + 8*s, col N = ln ----
#pragma unroll
  for (int mt = 0; mt < 2; ++mt) {
#pragma unroll
    for (int nt = 0; nt < 4; ++nt) {
      const int c = colBase + wn * 64 + nt * 16 + ln;
#pragma unroll
      for (int v = 0; v < 8; ++v) {
        const int r = rowBase + wm * 32 + mt * 16 + v + s * 8;
        if constexpr (OUT_F32_BIAS) {
          ((float*)Cptr)[(size_t)r * Nc + c] = acc[mt][nt][v] + bias[c];
        } else {
          ((_Float16*)Cptr)[(size_t)r * Nc + c] = (_Float16)acc[mt][nt][v];
        }
      }
    }
  }
}

// ---------------------------------------------------------------------------
// Flash-style attention. One block = (b, h, 128 Q rows); 8 waves x 16 rows.
// Streams J in chunks of 64 through LDS; online softmax in f32.
// ---------------------------------------------------------------------------
__global__ __launch_bounds__(256)
void attn_kernel(const _Float16* __restrict__ Qh, const _Float16* __restrict__ Kh,
                 const _Float16* __restrict__ Vh, _Float16* __restrict__ Oh)
{
  constexpr int JT = 64, LDT = 72;
  __shared__ __align__(16) _Float16 sK [64 * LDT];      // [j][d]
  __shared__ __align__(16) _Float16 sVt[64 * LDT];      // [d][j]
  __shared__ __align__(16) _Float16 sP [8 * 16 * LDT];  // per-wave 16x64

  const int tid = threadIdx.x, lane = tid & 31, wave = tid >> 5;
  const int s = lane >> 4, ln = lane & 15;

  const int bid     = blockIdx.x;
  const int rowTile = bid & 15;         // N/128
  const int h       = (bid >> 4) & 15;  // H
  const int b       = bid >> 8;         // B
  const int rowBase = rowTile * 128 + wave * 16;

  // Q A-fragments resident for the whole kernel (rows = rowBase + ln)
  const _Float16* qp = Qh + ((size_t)(b * Nseq + rowBase + ln)) * INNER + h * Dd;
  const v16h qf0 = ldfrag(qp + s * 8,      qp + 16 + s * 8);   // d 0..31
  const v16h qf1 = ldfrag(qp + 32 + s * 8, qp + 48 + s * 8);   // d 32..63

  float m_run[8], l_run[8];
  v8f oacc[4];
#pragma unroll
  for (int v = 0; v < 8; ++v) { m_run[v] = -3.0e38f; l_run[v] = 0.0f; }
#pragma unroll
  for (int dt = 0; dt < 4; ++dt) oacc[dt] = {};

  _Float16* sPw = sP + wave * 16 * LDT;

  for (int jc = 0; jc < Jseq / JT; ++jc) {
    __syncthreads();  // previous chunk's K/V/P reads complete
    // ---- load K tile [j][d] natural, V tile transposed [d][j] ----
    {
      const int j  = tid >> 2;
      const int dc = (tid & 3) * 16;
      const size_t rowOff = ((size_t)(b * Jseq + jc * JT + j)) * INNER + h * Dd + dc;
      const v8h klo = *(const v8h*)(Kh + rowOff);
      const v8h khi = *(const v8h*)(Kh + rowOff + 8);
      *(v8h*)(sK + j * LDT + dc)     = klo;
      *(v8h*)(sK + j * LDT + dc + 8) = khi;
      const v8h vlo = *(const v8h*)(Vh + rowOff);
      const v8h vhi = *(const v8h*)(Vh + rowOff + 8);
#pragma unroll
      for (int i = 0; i < 8; ++i) sVt[(dc + i) * LDT + j]     = vlo[i];
#pragma unroll
      for (int i = 0; i < 8; ++i) sVt[(dc + 8 + i) * LDT + j] = vhi[i];
    }
    __syncthreads();

    // ---- S = (Q K^T) * scale for 4 column tiles of 16 ----
    v8f st[4];
#pragma unroll
    for (int jt = 0; jt < 4; ++jt) {
      const _Float16* bp = sK + (jt * 16 + ln) * LDT + s * 8; // n=j_local, k=d
      const v16h b0 = ldfrag(bp,      bp + 16);      // d 0..31
      const v16h b1 = ldfrag(bp + 32, bp + 48);      // d 32..63
      v8f a = {};
      a = wmma32(qf0, b0, a);
      a = wmma32(qf1, b1, a);
#pragma unroll
      for (int v = 0; v < 8; ++v) a[v] *= 0.125f;    // D^-0.5
      st[jt] = a;
    }

    // ---- online softmax (row M = v + 8*s; reduce over 16-lane half) ----
    float alpha[8], rs[8];
#pragma unroll
    for (int v = 0; v < 8; ++v) {
      float m = fmaxf(fmaxf(st[0][v], st[1][v]), fmaxf(st[2][v], st[3][v]));
#pragma unroll
      for (int off = 1; off < 16; off <<= 1) m = fmaxf(m, __shfl_xor(m, off, 32));
      const float mnew = fmaxf(m_run[v], m);
      alpha[v] = __expf(m_run[v] - mnew);
      m_run[v] = mnew;
      rs[v] = 0.0f;
    }
#pragma unroll
    for (int jt = 0; jt < 4; ++jt) {
#pragma unroll
      for (int v = 0; v < 8; ++v) {
        const float p = __expf(st[jt][v] - m_run[v]);
        rs[v] += p;
        sPw[(v + s * 8) * LDT + jt * 16 + ln] = (_Float16)p;  // C->A relayout via LDS
      }
    }
#pragma unroll
    for (int v = 0; v < 8; ++v) {
      float r = rs[v];
#pragma unroll
      for (int off = 1; off < 16; off <<= 1) r += __shfl_xor(r, off, 32);
      l_run[v] = l_run[v] * alpha[v] + r;
#pragma unroll
      for (int dt = 0; dt < 4; ++dt) oacc[dt][v] *= alpha[v];
    }
    __syncthreads();  // P visible (cross-lane within wave)

    // ---- O += P(16x64) * V(64x64) ----
    const _Float16* pp = sPw + ln * LDT;
    const v16h pf0 = ldfrag(pp + s * 8,      pp + 16 + s * 8);   // k(j) 0..31
    const v16h pf1 = ldfrag(pp + 32 + s * 8, pp + 48 + s * 8);   // k(j) 32..63
#pragma unroll
    for (int dt = 0; dt < 4; ++dt) {
      const _Float16* vp = sVt + (dt * 16 + ln) * LDT + s * 8;  // n=d_local, k=j
      const v16h vf0 = ldfrag(vp,      vp + 16);
      const v16h vf1 = ldfrag(vp + 32, vp + 48);
      oacc[dt] = wmma32(pf0, vf0, oacc[dt]);
      oacc[dt] = wmma32(pf1, vf1, oacc[dt]);
    }
  }

  // ---- normalize and store f16 [B*N, INNER] ----
#pragma unroll
  for (int dt = 0; dt < 4; ++dt) {
    const int d = dt * 16 + ln;
#pragma unroll
    for (int v = 0; v < 8; ++v) {
      const int r = rowBase + v + s * 8;
      const float o = oacc[dt][v] / l_run[v];
      Oh[((size_t)(b * Nseq + r)) * INNER + h * Dd + d] = (_Float16)o;
    }
  }
}

// ---------------------------------------------------------------------------
extern "C" void kernel_launch(void* const* d_in, const int* in_sizes, int n_in,
                              void* d_out, int out_size, void* d_ws, size_t ws_size,
                              hipStream_t stream) {
  (void)in_sizes; (void)n_in; (void)out_size; (void)ws_size;
  const float* x   = (const float*)d_in[0];
  const float* ctx = (const float*)d_in[1];
  const float* Wq  = (const float*)d_in[2];
  const float* Wk  = (const float*)d_in[3];
  const float* Wv  = (const float*)d_in[4];
  const float* Wo  = (const float*)d_in[5];
  const float* bo  = (const float*)d_in[6];
  float* out = (float*)d_out;

  _Float16* Qh = (_Float16*)d_ws;                       // 16384x1024 f16
  _Float16* Kh = Qh + (size_t)Bsz * Nseq * INNER;       //  8192x1024 f16
  _Float16* Vh = Kh + (size_t)Bsz * Jseq * INNER;       //  8192x1024 f16
  _Float16* Oh = Vh + (size_t)Bsz * Jseq * INNER;       // 16384x1024 f16

  const dim3 blk(256);
  // projections (f32 in, f16 out)
  gemm_kernel<false, false><<<dim3(INNER / 128, (Bsz * Nseq) / 128), blk, 0, stream>>>(
      x, Wq, nullptr, Qh, Bsz * Nseq, QD, INNER);
  gemm_kernel<false, false><<<dim3(INNER / 128, (Bsz * Jseq) / 128), blk, 0, stream>>>(
      ctx, Wk, nullptr, Kh, Bsz * Jseq, CD, INNER);
  gemm_kernel<false, false><<<dim3(INNER / 128, (Bsz * Jseq) / 128), blk, 0, stream>>>(
      ctx, Wv, nullptr, Vh, Bsz * Jseq, CD, INNER);
  // attention
  attn_kernel<<<dim3(Bsz * Hh * (Nseq / 128)), blk, 0, stream>>>(Qh, Kh, Vh, Oh);
  // output projection (f16 A, f32 out + bias)
  gemm_kernel<true, true><<<dim3(QD / 128, (Bsz * Nseq) / 128), blk, 0, stream>>>(
      Oh, Wo, bo, out, Bsz * Nseq, INNER, QD);
}
